// MultiHeadAttention_712964571475
// MI455X (gfx1250) — compile-verified, matrix operand order checked by reading, not run on hardware
//
#include <hip/hip_runtime.h>
#include <hip/hip_bf16.h>

#define BATCH 2
#define SEQ   2048
#define DM    1024
#define NH    16
#define DK    64
#define MTOT  (BATCH * SEQ)   // 4096 rows in the "flattened token" dimension

typedef __bf16 bf16_t;
typedef __attribute__((ext_vector_type(16))) __bf16 v16bf;
typedef __attribute__((ext_vector_type(8)))  float  v8f;

// ---------------------------------------------------------------------------
// WMMA helper: D = A(16x32 bf16) x B(32x16 bf16) + C(16x16 f32)
// ---------------------------------------------------------------------------
static __device__ __forceinline__ v8f wmma_bf16(v16bf a, v16bf b, v8f c) {
  return __builtin_amdgcn_wmma_f32_16x16x32_bf16(
      /*neg_a=*/false, a, /*neg_b=*/false, b,
      /*c_mod=*/(short)0, c, /*reuse_a=*/false, /*reuse_b=*/false);
}

union TileU { uint4 u[2]; v16bf v; };

// Load a 16x32 bf16 operand tile (A operand, or B operand from an N-major,
// K-contiguous buffer). Lane layout per CDNA5 ISA (16-bit A 16x32):
//   lane&15 -> row, (lane>>4)*8 -> k base; per lane holds k = kb..kb+7 and
//   kb+16..kb+23 => two 16-byte loads.
static __device__ __forceinline__ v16bf load_tile_bf16(const bf16_t* __restrict__ base,
                                                       int row0, int k0, int ld, int lane) {
  const int r  = row0 + (lane & 15);
  const int kb = k0 + ((lane >> 4) << 3);
  const bf16_t* p = base + (size_t)r * ld + kb;
  TileU t;
  t.u[0] = *reinterpret_cast<const uint4*>(p);
  t.u[1] = *reinterpret_cast<const uint4*>(p + 16);
  return t.v;
}

// Same tile shape but sourced from f32 memory, converted to bf16 in-register.
static __device__ __forceinline__ v16bf load_tile_f32(const float* __restrict__ base,
                                                      int row0, int k0, int ld, int lane) {
  const int r  = row0 + (lane & 15);
  const int kb = k0 + ((lane >> 4) << 3);
  const float* p = base + (size_t)r * ld + kb;
  float4 f0 = *reinterpret_cast<const float4*>(p + 0);
  float4 f1 = *reinterpret_cast<const float4*>(p + 4);
  float4 f2 = *reinterpret_cast<const float4*>(p + 16);
  float4 f3 = *reinterpret_cast<const float4*>(p + 20);
  float f[16] = {f0.x, f0.y, f0.z, f0.w, f1.x, f1.y, f1.z, f1.w,
                 f2.x, f2.y, f2.z, f2.w, f3.x, f3.y, f3.z, f3.w};
  v16bf v;
#pragma unroll
  for (int i = 0; i < 16; ++i) v[i] = (bf16_t)f[i];
  return v;
}

// ---------------------------------------------------------------------------
// Conversion kernels
// ---------------------------------------------------------------------------
__global__ void k_cvt_scale(const float* __restrict__ src, bf16_t* __restrict__ dst,
                            int n, float scale) {
  int i = blockIdx.x * blockDim.x + threadIdx.x;
  if (i < n) dst[i] = (bf16_t)(src[i] * scale);
}

// WT[n][k] = W[k][n]  (1024x1024), so GEMM B-operand lanes read contiguous k.
__global__ void k_cvt_wT(const float* __restrict__ W, bf16_t* __restrict__ WT) {
  int i = blockIdx.x * blockDim.x + threadIdx.x;   // 1M threads
  int k = i & (DM - 1);
  int n = i >> 10;
  WT[i] = (bf16_t)W[(size_t)k * DM + n];
}

// ---------------------------------------------------------------------------
// Projection GEMM: C[MTOT, DM] = Xbf[MTOT, DM] @ WT^T   (WT stored N-major)
// One wave computes one 16-row M tile x four 16-col N tiles (64 cols).
// MODE 0: store bf16 row-major [MTOT, DM]   (Q, K)
// MODE 1: store per-head transposed Vt [B][NH][DK][SEQ]   (V)
// ---------------------------------------------------------------------------
template <int MODE>
__global__ void k_proj(const bf16_t* __restrict__ X, const bf16_t* __restrict__ WT,
                       bf16_t* __restrict__ Out) {
  const int lane = threadIdx.x & 31;
  const int wv   = (blockIdx.x << 3) | (threadIdx.x >> 5);
  const int ng   = wv & 15;        // 16 groups of 4 n-tiles -> 64 n-tiles
  const int mt   = wv >> 4;        // 256 m-tiles
  const int m0   = mt * 16;
  const int n0   = ng * 64;

  v8f acc[4] = {v8f{}, v8f{}, v8f{}, v8f{}};
  for (int k0 = 0; k0 < DM; k0 += 32) {
    __builtin_prefetch(X + (size_t)(m0 + (lane & 15)) * DM + k0 + 32, 0, 0);
    v16bf a = load_tile_bf16(X, m0, k0, DM, lane);
#pragma unroll
    for (int n = 0; n < 4; ++n) {
      v16bf b = load_tile_bf16(WT, n0 + n * 16, k0, DM, lane);
      acc[n] = wmma_bf16(a, b, acc[n]);
    }
  }

  const int half  = lane >> 4;
  if (MODE == 0) {
    const int rbase = m0 + (half << 3);
#pragma unroll
    for (int n = 0; n < 4; ++n) {
      const int c = n0 + n * 16 + (lane & 15);
#pragma unroll
      for (int i = 0; i < 8; ++i)
        Out[(size_t)(rbase + i) * DM + c] = (bf16_t)acc[n][i];
    }
  } else {
    // Transposed per-head store: Vt[b][h][dh][s]; per lane 8 consecutive s.
    const int b  = m0 >> 11;                       // m0 / SEQ
    const int s0 = (m0 & (SEQ - 1)) + (half << 3);
#pragma unroll
    for (int n = 0; n < 4; ++n) {
      const int d  = n0 + n * 16 + (lane & 15);    // feature 0..1023
      const int h  = d >> 6;
      const int dh = d & (DK - 1);
      bf16_t* dst = Out + (((size_t)b * NH + h) * DK + dh) * SEQ + s0;
      union { bf16_t b8[8]; uint4 u; } pk;
#pragma unroll
      for (int i = 0; i < 8; ++i) pk.b8[i] = (bf16_t)acc[n][i];
      *reinterpret_cast<uint4*>(dst) = pk.u;
    }
  }
}

// ---------------------------------------------------------------------------
// Fused scores + softmax. One wave owns 16 query rows of one (b,h).
// Pass 1: stream K tiles through WMMA, online (max,sum) per row slot.
// Cross-lane merge within 16-lane halves. Pass 2: recompute, normalize,
// write A (f32) exactly once to d_out.
// ---------------------------------------------------------------------------
__global__ void k_scores_softmax(const bf16_t* __restrict__ Q, const bf16_t* __restrict__ K,
                                 float* __restrict__ A) {
  const int lane = threadIdx.x & 31;
  const int wv   = (blockIdx.x << 3) | (threadIdx.x >> 5);
  const int qt   = wv & 127;       // SEQ/16 q-tiles
  const int bh   = wv >> 7;        // 0..31
  const int b    = bh >> 4;
  const int h    = bh & 15;

  const bf16_t* Qb = Q + (size_t)b * SEQ * DM + h * DK;
  const bf16_t* Kb = K + (size_t)b * SEQ * DM + h * DK;
  float* Ab = A + (size_t)bh * SEQ * SEQ;

  const int q0 = qt * 16;
  const v16bf a0 = load_tile_bf16(Qb, q0, 0,  DM, lane);
  const v16bf a1 = load_tile_bf16(Qb, q0, 32, DM, lane);

  float m[8], l[8];
#pragma unroll
  for (int i = 0; i < 8; ++i) { m[i] = -1e30f; l[i] = 0.0f; }

  // ---- pass 1: running row max / sum ----
  for (int kt = 0; kt < SEQ / 16; ++kt) {
    v16bf b0 = load_tile_bf16(Kb, kt * 16, 0,  DM, lane);
    v16bf b1 = load_tile_bf16(Kb, kt * 16, 32, DM, lane);
    v8f c = v8f{};
    c = wmma_bf16(a0, b0, c);
    c = wmma_bf16(a1, b1, c);
#pragma unroll
    for (int i = 0; i < 8; ++i) {
      float s  = c[i];
      float mn = fmaxf(m[i], s);
      l[i] = l[i] * __expf(m[i] - mn) + __expf(s - mn);
      m[i] = mn;
    }
  }

  // ---- merge across the 16 lanes holding each row (xor within half-wave) ----
#pragma unroll
  for (int step = 1; step < 16; step <<= 1) {
#pragma unroll
    for (int i = 0; i < 8; ++i) {
      float m2 = __shfl_xor(m[i], step, 32);
      float l2 = __shfl_xor(l[i], step, 32);
      float mn = fmaxf(m[i], m2);
      l[i] = l[i] * __expf(m[i] - mn) + l2 * __expf(m2 - mn);
      m[i] = mn;
    }
  }
  float inv[8];
#pragma unroll
  for (int i = 0; i < 8; ++i) inv[i] = 1.0f / l[i];

  // ---- pass 2: recompute, normalize, single write of A ----
  const int rbase = q0 + ((lane >> 4) << 3);
  for (int kt = 0; kt < SEQ / 16; ++kt) {
    v16bf b0 = load_tile_bf16(Kb, kt * 16, 0,  DM, lane);
    v16bf b1 = load_tile_bf16(Kb, kt * 16, 32, DM, lane);
    v8f c = v8f{};
    c = wmma_bf16(a0, b0, c);
    c = wmma_bf16(a1, b1, c);
    const int col = kt * 16 + (lane & 15);
#pragma unroll
    for (int i = 0; i < 8; ++i) {
      float p = __expf(c[i] - m[i]) * inv[i];
      Ab[(size_t)(rbase + i) * SEQ + col] = p;
    }
  }
}

// ---------------------------------------------------------------------------
// H_heads = A @ V : per (b,h), [SEQ,SEQ] x [SEQ,DK]. A read once from d_out
// (f32 -> bf16 in-register); Vt is N-major K-contiguous. One wave: one q-tile
// x all 4 n-tiles (full d_k). Output written into Hcat [MTOT, DM] bf16.
// ---------------------------------------------------------------------------
__global__ void k_av(const float* __restrict__ A, const bf16_t* __restrict__ Vt,
                     bf16_t* __restrict__ Hcat) {
  const int lane = threadIdx.x & 31;
  const int wv   = (blockIdx.x << 3) | (threadIdx.x >> 5);
  const int qt   = wv & 127;
  const int bh   = wv >> 7;
  const int b    = bh >> 4;
  const int h    = bh & 15;

  const float*  Ab = A  + (size_t)bh * SEQ * SEQ;
  const bf16_t* Vb = Vt + (size_t)bh * DK * SEQ;

  v8f acc[4] = {v8f{}, v8f{}, v8f{}, v8f{}};
  for (int k0 = 0; k0 < SEQ; k0 += 32) {
    v16bf a = load_tile_f32(Ab, qt * 16, k0, SEQ, lane);
#pragma unroll
    for (int n = 0; n < 4; ++n) {
      v16bf bt = load_tile_bf16(Vb, n * 16, k0, SEQ, lane);
      acc[n] = wmma_bf16(a, bt, acc[n]);
    }
  }

  const int rbase = qt * 16 + ((lane >> 4) << 3);
  const size_t rowb = (size_t)b * SEQ;
#pragma unroll
  for (int n = 0; n < 4; ++n) {
    const int c = h * DK + n * 16 + (lane & 15);
#pragma unroll
    for (int i = 0; i < 8; ++i)
      Hcat[(rowb + rbase + i) * DM + c] = (bf16_t)acc[n][i];
  }
}

// ---------------------------------------------------------------------------
// Output projection: H = Hcat @ W_h + b_h  (f32 out to d_out).
// ---------------------------------------------------------------------------
__global__ void k_outproj(const bf16_t* __restrict__ Hc, const bf16_t* __restrict__ WT,
                          const float* __restrict__ bias, float* __restrict__ Out) {
  const int lane = threadIdx.x & 31;
  const int wv   = (blockIdx.x << 3) | (threadIdx.x >> 5);
  const int ng   = wv & 15;
  const int mt   = wv >> 4;
  const int m0   = mt * 16;
  const int n0   = ng * 64;

  v8f acc[4] = {v8f{}, v8f{}, v8f{}, v8f{}};
  for (int k0 = 0; k0 < DM; k0 += 32) {
    v16bf a = load_tile_bf16(Hc, m0, k0, DM, lane);
#pragma unroll
    for (int n = 0; n < 4; ++n) {
      v16bf b = load_tile_bf16(WT, n0 + n * 16, k0, DM, lane);
      acc[n] = wmma_bf16(a, b, acc[n]);
    }
  }

  const int rbase = m0 + ((lane >> 4) << 3);
#pragma unroll
  for (int n = 0; n < 4; ++n) {
    const int c = n0 + n * 16 + (lane & 15);
    const float bv = bias[c];
#pragma unroll
    for (int i = 0; i < 8; ++i)
      Out[(size_t)(rbase + i) * DM + c] = acc[n][i] + bv;
  }
}

// ---------------------------------------------------------------------------
// Host launcher
// ---------------------------------------------------------------------------
extern "C" void kernel_launch(void* const* d_in, const int* in_sizes, int n_in,
                              void* d_out, int out_size, void* d_ws, size_t ws_size,
                              hipStream_t stream) {
  (void)in_sizes; (void)n_in; (void)out_size; (void)ws_size;
  const float* Xq = (const float*)d_in[0];
  const float* Xk = (const float*)d_in[1];
  const float* Xv = (const float*)d_in[2];
  const float* Wq = (const float*)d_in[3];
  const float* Wk = (const float*)d_in[4];
  const float* Wv = (const float*)d_in[5];
  const float* Wh = (const float*)d_in[6];
  const float* bh = (const float*)d_in[7];

  float* Hout = (float*)d_out;                              // [B,S,DM]
  float* Aout = Hout + (size_t)MTOT * DM;                   // [B,H,S,S]

  const size_t NX = (size_t)MTOT * DM;   // 4,194,304
  const size_t NW = (size_t)DM * DM;     // 1,048,576
  bf16_t* w   = (bf16_t*)d_ws;
  bf16_t* Xqb = w;
  bf16_t* Xkb = Xqb + NX;
  bf16_t* Xvb = Xkb + NX;
  bf16_t* WqT = Xvb + NX;
  bf16_t* WkT = WqT + NW;
  bf16_t* WvT = WkT + NW;
  bf16_t* WhT = WvT + NW;
  bf16_t* Qb  = WhT + NW;
  bf16_t* Kb  = Qb + NX;
  bf16_t* Vt  = Kb + NX;   // [B][NH][DK][SEQ]
  bf16_t* Hc  = Vt + NX;   // [MTOT, DM]
  // total: 3*NX + 4*NW + 4*NX bf16 = 64 MB

  const dim3 blk(256);
  const int cvt_blocks = (int)((NX + 255) / 256);
  const float inv_sqrt_dk = 0.125f;  // 1/sqrt(64), folded into X_q

  k_cvt_scale<<<cvt_blocks, blk, 0, stream>>>(Xq, Xqb, (int)NX, inv_sqrt_dk);
  k_cvt_scale<<<cvt_blocks, blk, 0, stream>>>(Xk, Xkb, (int)NX, 1.0f);
  k_cvt_scale<<<cvt_blocks, blk, 0, stream>>>(Xv, Xvb, (int)NX, 1.0f);

  const int wT_blocks = (int)(NW / 256);
  k_cvt_wT<<<wT_blocks, blk, 0, stream>>>(Wq, WqT);
  k_cvt_wT<<<wT_blocks, blk, 0, stream>>>(Wk, WkT);
  k_cvt_wT<<<wT_blocks, blk, 0, stream>>>(Wv, WvT);
  k_cvt_wT<<<wT_blocks, blk, 0, stream>>>(Wh, WhT);

  // 4096 waves / 8 waves per 256-thread block = 512 blocks for each stage
  k_proj<0><<<512, blk, 0, stream>>>(Xqb, WqT, Qb);
  k_proj<0><<<512, blk, 0, stream>>>(Xkb, WkT, Kb);
  k_proj<1><<<512, blk, 0, stream>>>(Xvb, WvT, Vt);

  k_scores_softmax<<<512, blk, 0, stream>>>(Qb, Kb, Aout);
  k_av<<<512, blk, 0, stream>>>(Aout, Vt, Hc);
  k_outproj<<<512, blk, 0, stream>>>(Hc, WhT, bh, Hout);
}
